// HyperMP_33852932227669
// MI455X (gfx1250) — matrix operand order repaired and checked by reading, not compiled, not run on hardware
//
#include <hip/hip_runtime.h>
#include <cstdint>

// ---------------------------------------------------------------------------
// Problem constants (match reference)
// ---------------------------------------------------------------------------
#define Bc   4
#define NMc  30000
#define NRc  60000
#define Ec   100000
#define INc  32
#define Hc   128
#define OUTc 32
#define Lc   2
#define EPSc 1e-5f

typedef __bf16 bf16_t;
typedef bf16_t v16bf __attribute__((ext_vector_type(16)));
typedef float  v8f   __attribute__((ext_vector_type(8)));
typedef unsigned short ushort_t;   // raw bf16 storage

union Frag {
    uint4 u[2];
    v16bf v;
};

__device__ __forceinline__ ushort_t f2bf(float f) {
    union { float f; unsigned u; } x;
    x.f = f;
    unsigned r = x.u + 0x7FFFu + ((x.u >> 16) & 1u);   // round-to-nearest-even
    return (ushort_t)(r >> 16);
}
__device__ __forceinline__ float bf2f(ushort_t u) {
    union { unsigned u; float f; } x;
    x.u = ((unsigned)u) << 16;
    return x.f;
}

// gfx1250 async copy: 16B global -> LDS, tracked by ASYNCcnt. Per-lane LDS
// offset VGPR lets us land in the padded (bank-conflict-free) LDS layout.
__device__ __forceinline__ void async_cp16(unsigned lds_off, const void* gp) {
    asm volatile("global_load_async_to_lds_b128 %0, %1, off"
                 :: "v"(lds_off), "v"(gp) : "memory");
}
__device__ __forceinline__ void async_wait0() {
    asm volatile("s_wait_asynccnt 0x0" ::: "memory");
}

// ---------------------------------------------------------------------------
// Fused WMMA GEMM:  Y = epi( (X [+X2]) @ W + bias )
//   Wt: pre-transposed bf16 weights, layout [N][K] row-contiguous.
//   TIN/TOUT in {float, ushort_t(bf16)}.  X:[R,K], Y:[R,N], R % 64 == 0.
//   128 threads = 4 waves, 64 rows/block, full N per block.
//   bf16 inputs stage via global_load_async_to_lds_b128; f32 inputs convert.
//   Compute: v_wmma_f32_16x16x32_bf16 (f32 accumulate).
// ---------------------------------------------------------------------------
template <int KS, int NT, bool RELU, bool BN, bool RESID, bool ADDIN, bool DUP,
          typename TIN, typename TOUT>
__global__ __launch_bounds__(128) void gemm_kernel(
    const TIN* __restrict__ X, const TIN* __restrict__ X2,
    const ushort_t* __restrict__ Wt, const float* __restrict__ bias,
    const float* __restrict__ scale, const float* __restrict__ shift,
    TOUT* __restrict__ Y, TOUT* __restrict__ Y2, int R)
{
    constexpr int K  = KS * 32;
    constexpr int N  = NT * 16;
    constexpr int SK = K + 8;                 // padded LDS stride (bf16 units)
    constexpr bool IN_BF  = (sizeof(TIN)  == 2);
    constexpr bool OUT_BF = (sizeof(TOUT) == 2);

    __shared__ __align__(16) ushort_t lW[N * SK];    // W^T  [n][k]
    __shared__ __align__(16) ushort_t lX[64 * SK];   // X tile [r][k]

    const int tid     = threadIdx.x;
    const int rowBase = blockIdx.x * 64;

    // --- stage W^T (bf16, contiguous) via async copy into padded LDS ---
    {
        constexpr int CPR = K / 8;                   // 16B chunks per row
        for (int c = tid; c < N * CPR; c += 128) {
            int n = c / CPR, kc = c - n * CPR;
            async_cp16((unsigned)(uintptr_t)&lW[n * SK + kc * 8],
                       Wt + (size_t)n * K + kc * 8);
        }
    }
    // --- stage X tile ---
    if constexpr (IN_BF && !ADDIN) {
        constexpr int CPR = K / 8;
        for (int c = tid; c < 64 * CPR; c += 128) {
            int r = c / CPR, kc = c - r * CPR;
            async_cp16((unsigned)(uintptr_t)&lX[r * SK + kc * 8],
                       (const ushort_t*)X + (size_t)(rowBase + r) * K + kc * 8);
        }
    } else {
        for (int i = tid; i < 64 * K; i += 128) {
            int r = i / K, k = i - r * K;
            size_t gi = (size_t)(rowBase + r) * K + k;
            float v;
            if constexpr (IN_BF) {
                v = bf2f(X[gi]);
                if (ADDIN) v += bf2f(X2[gi]);
            } else {
                v = X[gi];
                if (ADDIN) v += X2[gi];
            }
            lX[r * SK + k] = f2bf(v);
        }
    }
    async_wait0();
    __syncthreads();

    const int wave = tid >> 5;
    const int lane = tid & 31;
    const int hi   = lane >> 4;
    const int ln   = lane & 15;

    v8f acc[NT];
    const v8f zero8 = {0.f, 0.f, 0.f, 0.f, 0.f, 0.f, 0.f, 0.f};
#pragma unroll
    for (int nt = 0; nt < NT; ++nt) acc[nt] = zero8;

#pragma unroll
    for (int ks = 0; ks < KS; ++ks) {
        // A fragment: row = ln; K chunks {0-7,16-23} (hi=0) / {8-15,24-31} (hi=1)
        Frag a;
        const int abase = (wave * 16 + ln) * SK + ks * 32 + hi * 8;
        a.u[0] = *(const uint4*)&lX[abase];
        a.u[1] = *(const uint4*)&lX[abase + 16];
#pragma unroll
        for (int nt = 0; nt < NT; ++nt) {
            // B fragment: col = ln; K chunks {0-7,16-23} (hi=0) / {8-15,24-31} (hi=1)
            Frag b;
            const int bbase = (nt * 16 + ln) * SK + ks * 32 + hi * 8;
            b.u[0] = *(const uint4*)&lW[bbase];
            b.u[1] = *(const uint4*)&lW[bbase + 16];
            acc[nt] = __builtin_amdgcn_wmma_f32_16x16x32_bf16(
                false, a.v, false, b.v, (short)0, acc[nt], false, false);
        }
    }

    // --- epilogue: bias -> (ReLU) -> (affine BN) -> (residual) -> store ---
#pragma unroll
    for (int nt = 0; nt < NT; ++nt) {
        const int n  = nt * 16 + ln;
        const float bs = bias[n];
        const float sc = BN ? scale[n] : 1.f;
        const float sh = BN ? shift[n] : 0.f;
#pragma unroll
        for (int r = 0; r < 8; ++r) {
            const int row = rowBase + wave * 16 + hi * 8 + r;
            float v = acc[nt][r] + bs;
            if (RELU) v = fmaxf(v, 0.f);
            if (BN)   v = v * sc + sh;
            const size_t oi = (size_t)row * N + n;
            if constexpr (OUT_BF) {
                if (RESID) v += bf2f(Y[oi]);
                ushort_t o = f2bf(v);
                Y[oi] = o;
                if (DUP) Y2[oi] = o;
            } else {
                if (RESID) v += Y[oi];
                Y[oi] = v;
                if (DUP) Y2[oi] = v;
            }
        }
    }
}

// ---------------------------------------------------------------------------
// Utility kernels
// ---------------------------------------------------------------------------
__global__ void k_zero4(float4* __restrict__ p, long n4) {
    long i = (long)blockIdx.x * blockDim.x + threadIdx.x;
    if (i < n4) p[i] = make_float4(0.f, 0.f, 0.f, 0.f);
}

__global__ void k_count(const int* __restrict__ dst, float* __restrict__ cnt, int E) {
    int e = blockIdx.x * blockDim.x + threadIdx.x;
    if (e < E) atomicAdd(&cnt[dst[e]], 1.0f);
}

__global__ void k_invert(float* __restrict__ c, int n) {
    int i = blockIdx.x * blockDim.x + threadIdx.x;
    if (i < n) c[i] = 1.0f / fmaxf(c[i], 1.0f);
}

// fold eval-BatchNorm into scale/shift: s = g*rsqrt(var+eps); c = be - mu*s
__global__ void k_bnprep(const float* __restrict__ g, const float* __restrict__ var,
                         const float* __restrict__ be, const float* __restrict__ mu,
                         float* __restrict__ s, float* __restrict__ c, int n) {
    int i = blockIdx.x * blockDim.x + threadIdx.x;
    if (i < n) {
        float sv = g[i] * rsqrtf(var[i] + EPSc);
        s[i] = sv;
        c[i] = be[i] - mu[i] * sv;
    }
}

// weight prep: Wt[n*K + k] = bf16(W[k*N + n])  (transpose + f32->bf16, once/launch)
__global__ void k_wprep(const float* __restrict__ W, ushort_t* __restrict__ Wt,
                        int K, int N) {
    int i = blockIdx.x * blockDim.x + threadIdx.x;
    if (i < K * N) {
        int n = i / K, k = i - n * K;
        Wt[i] = f2bf(W[(size_t)k * N + n]);
    }
}

// Pre-scaled scatter-mean: out[b,dst[e],:] += bf2f(X[b,src[e],:]) * inv[dst[e]]
// bf16 gather reads (half traffic), fp32 atomics into L2-resident agg buffer.
__global__ __launch_bounds__(256) void k_scatter(
    const ushort_t* __restrict__ X, const int* __restrict__ src,
    const int* __restrict__ dst, const float* __restrict__ inv,
    float* __restrict__ out, int E, int n_src, int n_dst)
{
    const int row = blockIdx.x * 8 + (threadIdx.x >> 5);
    if (row >= E * Bc) return;
    const int b = row / E;
    const int e = row - b * E;
    const int s = src[e];
    const int d = dst[e];
    const float w = inv[d];
    const int lane = threadIdx.x & 31;
    const uint2 u = ((const uint2*)(X + ((size_t)b * n_src + s) * Hc))[lane];
    float* op = out + ((size_t)b * n_dst + d) * Hc + lane * 4;
    atomicAdd(op + 0, bf2f((ushort_t)(u.x & 0xffff)) * w);
    atomicAdd(op + 1, bf2f((ushort_t)(u.x >> 16))    * w);
    atomicAdd(op + 2, bf2f((ushort_t)(u.y & 0xffff)) * w);
    atomicAdd(op + 3, bf2f((ushort_t)(u.y >> 16))    * w);
}

// ---------------------------------------------------------------------------
// Host-side orchestration
// ---------------------------------------------------------------------------
static inline void zero_f(float* p, size_t n, hipStream_t st) {
    long n4 = (long)(n / 4);
    k_zero4<<<(unsigned)((n4 + 255) / 256), 256, 0, st>>>((float4*)p, n4);
}

extern "C" void kernel_launch(void* const* d_in, const int* in_sizes, int n_in,
                              void* d_out, int out_size, void* d_ws, size_t ws_size,
                              hipStream_t stream) {
    const float* node_inputs = (const float*)d_in[0];
    const int*   r2e = (const int*)d_in[1];
    const int*   p2e = (const int*)d_in[2];
    const int*   e2r = (const int*)d_in[3];
    const int*   e2p = (const int*)d_in[4];
    const float* Win  = (const float*)d_in[5];
    const float* b_in = (const float*)d_in[6];
    const float* Wf[8] = { (const float*)d_in[7],  (const float*)d_in[8],
                           (const float*)d_in[9],  (const float*)d_in[10],
                           (const float*)d_in[11], (const float*)d_in[12],
                           (const float*)d_in[13], (const float*)d_in[14] };
    const float* b_r2r = (const float*)d_in[15];
    const float* b_p2r = (const float*)d_in[16];
    const float* b_ra1 = (const float*)d_in[17];
    const float* b_ra2 = (const float*)d_in[18];
    const float* b_rr  = (const float*)d_in[19];
    const float* b_rp  = (const float*)d_in[20];
    const float* b_ma1 = (const float*)d_in[21];
    const float* b_ma2 = (const float*)d_in[22];
    const float* be_r  = (const float*)d_in[23];
    const float* mu_r  = (const float*)d_in[24];
    const float* be_m  = (const float*)d_in[25];
    const float* mu_m  = (const float*)d_in[26];
    const float* g_r   = (const float*)d_in[27];
    const float* g_m   = (const float*)d_in[28];
    const float* var_r = (const float*)d_in[29];
    const float* var_m = (const float*)d_in[30];
    const float* Wo1 = (const float*)d_in[31];
    const float* bo1 = (const float*)d_in[32];
    const float* Wo2 = (const float*)d_in[33];
    const float* bo2 = (const float*)d_in[34];
    float* out = (float*)d_out;

    const int RM = Bc * NMc;   // 120000 rows (% 64 == 0)
    const int RR = Bc * NRc;   // 240000 rows (% 64 == 0)
    const size_t SzM = (size_t)RM * Hc;
    const size_t SzR = (size_t)RR * Hc;
    const int HH = Hc * Hc;

    // workspace layout: regions sized in f32, some used as bf16
    float* ws  = (float*)d_ws;
    float* h   = ws;                 // bf16 use
    float* h0  = h  + SzM;           // bf16 use
    float* mA  = h0 + SzM;           // bf16 (rh/tmp) and f32 (agg2) use
    float* mB  = mA + SzM;           // bf16 use
    float* r0  = mB + SzM;           // f32 (agg) and bf16 (rxn) use
    float* r1  = r0 + SzR;           // bf16 use
    float* inv_r2e = r1 + SzR;
    float* inv_p2e = inv_r2e + NRc;
    float* inv_e2r = inv_p2e + NRc;
    float* inv_e2p = inv_e2r + NMc;
    float* s_r = inv_e2p + NMc;
    float* c_r = s_r + Lc * Hc;
    float* s_m = c_r + Lc * Hc;
    float* c_m = s_m + Lc * Hc;
    // pre-transposed bf16 weights (~573 KB)
    ushort_t* wtb   = (ushort_t*)(c_m + Lc * Hc);
    ushort_t* Win_t = wtb;             wtb += INc * Hc;
    ushort_t* Wt[8];
    for (int t = 0; t < 8; ++t) { Wt[t] = wtb; wtb += (size_t)Lc * HH; }
    ushort_t* Wo1_t = wtb;             wtb += HH;
    ushort_t* Wo2_t = wtb;

    ushort_t* h_b  = (ushort_t*)h;
    ushort_t* h0_b = (ushort_t*)h0;
    ushort_t* mA_b = (ushort_t*)mA;
    ushort_t* mB_b = (ushort_t*)mB;
    ushort_t* r0_b = (ushort_t*)r0;
    ushort_t* r1_b = (ushort_t*)r1;

    // --- per-launch prep: weight transpose+cvt, BN folds, inverse counts ---
    k_wprep<<<(INc * Hc + 255) / 256, 256, 0, stream>>>(Win, Win_t, INc, Hc);
    for (int t = 0; t < 8; ++t)
        for (int l = 0; l < Lc; ++l)
            k_wprep<<<(HH + 255) / 256, 256, 0, stream>>>(
                Wf[t] + (size_t)l * HH, Wt[t] + (size_t)l * HH, Hc, Hc);
    k_wprep<<<(HH + 255) / 256, 256, 0, stream>>>(Wo1, Wo1_t, Hc, Hc);
    k_wprep<<<(Hc * OUTc + 255) / 256, 256, 0, stream>>>(Wo2, Wo2_t, Hc, OUTc);

    k_bnprep<<<1, Lc * Hc, 0, stream>>>(g_r, var_r, be_r, mu_r, s_r, c_r, Lc * Hc);
    k_bnprep<<<1, Lc * Hc, 0, stream>>>(g_m, var_m, be_m, mu_m, s_m, c_m, Lc * Hc);
    zero_f(inv_r2e, 2 * NRc + 2 * NMc, stream);
    k_count<<<(Ec + 255) / 256, 256, 0, stream>>>(r2e + Ec, inv_r2e, Ec);
    k_count<<<(Ec + 255) / 256, 256, 0, stream>>>(p2e + Ec, inv_p2e, Ec);
    k_count<<<(Ec + 255) / 256, 256, 0, stream>>>(e2r + Ec, inv_e2r, Ec);
    k_count<<<(Ec + 255) / 256, 256, 0, stream>>>(e2p + Ec, inv_e2p, Ec);
    k_invert<<<(2 * NRc + 2 * NMc + 255) / 256, 256, 0, stream>>>(
        inv_r2e, 2 * NRc + 2 * NMc);

    const int scat_blocks = (Bc * Ec + 7) / 8;

    // --- input projection: h = h0 = node_inputs @ Win + b_in  (K=32) ---
    gemm_kernel<1, 8, false, false, false, false, true, float, ushort_t>
        <<<RM / 64, 128, 0, stream>>>(node_inputs, nullptr, Win_t, b_in,
                                      nullptr, nullptr, h_b, h0_b, RM);

    for (int l = 0; l < Lc; ++l) {
        const size_t wo = (size_t)l * HH;
        const size_t bo = (size_t)l * Hc;

        // rh = h@W_r2r+b ; ph = h@W_p2r+b   (bf16 -> bf16)
        gemm_kernel<4, 8, false, false, false, false, false, ushort_t, ushort_t>
            <<<RM / 64, 128, 0, stream>>>(h_b, nullptr, Wt[0] + wo, b_r2r + bo,
                                          nullptr, nullptr, mA_b, nullptr, RM);
        gemm_kernel<4, 8, false, false, false, false, false, ushort_t, ushort_t>
            <<<RM / 64, 128, 0, stream>>>(h_b, nullptr, Wt[1] + wo, b_p2r + bo,
                                          nullptr, nullptr, mB_b, nullptr, RM);

        // agg = mean-scatter(rh via r2e) + mean-scatter(ph via p2e) -> r0 (f32)
        zero_f(r0, SzR, stream);
        k_scatter<<<scat_blocks, 256, 0, stream>>>(mA_b, r2e, r2e + Ec, inv_r2e,
                                                   r0, Ec, NMc, NRc);
        k_scatter<<<scat_blocks, 256, 0, stream>>>(mB_b, p2e, p2e + Ec, inv_p2e,
                                                   r0, Ec, NMc, NRc);

        // t = BN(relu(agg@W_ra1+b)) -> r1 (f32 in, bf16 out)
        gemm_kernel<4, 8, true, true, false, false, false, float, ushort_t>
            <<<RR / 64, 128, 0, stream>>>(r0, nullptr, Wt[2] + wo, b_ra1 + bo,
                                          s_r + bo, c_r + bo, r1_b, nullptr, RR);
        // rxn = t@W_ra2+b -> r0 region (bf16)
        gemm_kernel<4, 8, false, false, false, false, false, ushort_t, ushort_t>
            <<<RR / 64, 128, 0, stream>>>(r1_b, nullptr, Wt[3] + wo, b_ra2 + bo,
                                          nullptr, nullptr, r0_b, nullptr, RR);

        // rh2 -> r1 ; scatter ; ph2 -> r1 ; scatter (into mA f32)
        zero_f(mA, SzM, stream);
        gemm_kernel<4, 8, false, false, false, false, false, ushort_t, ushort_t>
            <<<RR / 64, 128, 0, stream>>>(r0_b, nullptr, Wt[4] + wo, b_rr + bo,
                                          nullptr, nullptr, r1_b, nullptr, RR);
        k_scatter<<<scat_blocks, 256, 0, stream>>>(r1_b, e2r, e2r + Ec, inv_e2r,
                                                   mA, Ec, NRc, NMc);
        gemm_kernel<4, 8, false, false, false, false, false, ushort_t, ushort_t>
            <<<RR / 64, 128, 0, stream>>>(r0_b, nullptr, Wt[5] + wo, b_rp + bo,
                                          nullptr, nullptr, r1_b, nullptr, RR);
        k_scatter<<<scat_blocks, 256, 0, stream>>>(r1_b, e2p, e2p + Ec, inv_e2p,
                                                   mA, Ec, NRc, NMc);

        // t2 = BN(relu(agg2@W_ma1+b)) -> mB ; h += t2@W_ma2+b (residual)
        gemm_kernel<4, 8, true, true, false, false, false, float, ushort_t>
            <<<RM / 64, 128, 0, stream>>>(mA, nullptr, Wt[6] + wo, b_ma1 + bo,
                                          s_m + bo, c_m + bo, mB_b, nullptr, RM);
        gemm_kernel<4, 8, false, false, true, false, false, ushort_t, ushort_t>
            <<<RM / 64, 128, 0, stream>>>(mB_b, nullptr, Wt[7] + wo, b_ma2 + bo,
                                          nullptr, nullptr, h_b, nullptr, RM);
    }

    // head: tmp = relu((h+h0)@Wo1+bo1) -> mA (bf16) ; out = tmp@Wo2+bo2 (f32)
    gemm_kernel<4, 8, true, false, false, true, false, ushort_t, ushort_t>
        <<<RM / 64, 128, 0, stream>>>(h_b, h0_b, Wo1_t, bo1,
                                      nullptr, nullptr, mA_b, nullptr, RM);
    gemm_kernel<4, 2, false, false, false, false, false, ushort_t, float>
        <<<RM / 64, 128, 0, stream>>>(mA_b, nullptr, Wo2_t, bo2,
                                      nullptr, nullptr, out, nullptr, RM);
}
